// FullyConnectedTensorProduct_59914793779560
// MI455X (gfx1250) — matrix operand order repaired and checked by reading, not compiled, not run on hardware
//
#include <hip/hip_runtime.h>

// ---------------------------------------------------------------------------
// FullyConnectedTensorProduct (64x0e+64x1o+64x2e)^2 -> 64x0e+64x1o+64x2e
// B=4096. Per path p: out[b,w,k] += sum_{u,v} ws[p][u,v,w] * Z_p[b,(u,v),k]
//   Z_p[b,(u,v),k] = sum_{i,j} C_p[i,j,k] x1[b,u,i] x2[b,v,j]
// v_wmma_f32_16x16x32_f16: A = W^T (16 w x 32 uv, f16), B = Z (32 uv x 16
// cols, f16), cols = (b_local + 16*k), f32 accumulation, K looped over uv.
// Weight slices streamed into LDS by the Tensor Data Mover (ping-pong),
// waited with s_wait_tensorcnt.
// ---------------------------------------------------------------------------

typedef __attribute__((ext_vector_type(16))) _Float16 v16h;
typedef __attribute__((ext_vector_type(8)))  _Float16 v8h;
typedef __attribute__((ext_vector_type(2)))  _Float16 h2v;
typedef __attribute__((ext_vector_type(8)))  float    v8f;
typedef __attribute__((ext_vector_type(4)))  unsigned int v4u;
typedef __attribute__((ext_vector_type(8)))  int      v8i;
typedef __attribute__((ext_vector_type(4)))  int      v4i;

#if defined(__HIP_DEVICE_COMPILE__) && defined(__gfx1250__) && \
    __has_builtin(__builtin_amdgcn_tensor_load_to_lds)
#define HAVE_TDM 1
#else
#define HAVE_TDM 0
#endif

#if HAVE_TDM
#if __clang_major__ >= 23
#define TDM_LOAD(g0, g1, g2, g3) \
  __builtin_amdgcn_tensor_load_to_lds((g0), (g1), (g2), (g3), \
                                      (v8i){0, 0, 0, 0, 0, 0, 0, 0}, 0)
#else
#define TDM_LOAD(g0, g1, g2, g3) \
  __builtin_amdgcn_tensor_load_to_lds((g0), (g1), (g2), (g3), 0)
#endif

// 1-D TDM descriptor: copy 4096 x 2-byte elements (8 KB) global -> LDS.
__device__ __forceinline__ void tdm_load_w(const _Float16* gsrc,
                                           unsigned int ldsOff) {
  unsigned long long ga = (unsigned long long)(uintptr_t)gsrc;
  v4u g0 = {1u,                                  // count=1, is_restore=0
            ldsOff,                              // lds_addr
            (unsigned int)ga,                    // global_addr[31:0]
            (unsigned int)((ga >> 32) & 0x01FFFFFFull) | 0x80000000u};  // type=2
  v8i g1 = {0x00010000,   // workgroup_mask=0, data_size=1 (2B)
            0x10000000,   // tensor_dim0[15:0]=4096 in bits 79:48 -> <<16
            0x00010000,   // tensor_dim1=1 in bits 111:80 -> <<16
            0x10000000,   // tile_dim0=4096 in bits 127:112 -> <<16
            0,            // tile_dim1=0 (unused), tile_dim2=0
            4096,         // tensor_dim0_stride[31:0]
            0, 0};
  v4i g2 = {0, 0, 0, 0};
  v4i g3 = {0, 0, 0, 0};
  TDM_LOAD(g0, g1, g2, g3);
}
#endif

// ---- Wigner-3j (real basis, unit Frobenius norm) as dense [11][5][5][5] ----
struct CGTab { float v[11][5][5][5]; };

static constexpr void sym3(CGTab& t, int p, int a, int b, int c, float val) {
  t.v[p][a][b][c] = val; t.v[p][a][c][b] = val; t.v[p][b][a][c] = val;
  t.v[p][b][c][a] = val; t.v[p][c][a][b] = val; t.v[p][c][b][a] = val;
}

static constexpr CGTab buildCG() {
  CGTab t{};
  constexpr float R3 = 0.57735026918962576f;   // 1/sqrt(3)
  constexpr float R5 = 0.44721359549995794f;   // 1/sqrt(5)
  constexpr float A  = 0.31622776601683794f;   // 1/sqrt(10)
  constexpr float B2 = 0.36514837167011072f;   // 2/sqrt(30)
  constexpr float B1 = 0.18257418583505536f;   // 1/sqrt(30)
  constexpr float C1 = 0.23904572186687872f;   // sqrt(2/35)
  constexpr float C2 = 0.11952286093343936f;   // sqrt(1/70)
  constexpr float C3 = 0.20701966780270626f;   // sqrt(3/70)
  t.v[0][0][0][0] = 1.0f;                                   // p0 (0,0,0)
  for (int j = 0; j < 3; j++) t.v[1][0][j][j] = R3;         // p1 (0,1,1)
  for (int j = 0; j < 5; j++) t.v[2][0][j][j] = R5;         // p2 (0,2,2)
  for (int i = 0; i < 3; i++) t.v[3][i][0][i] = R3;         // p3 (1,0,1)
  for (int i = 0; i < 3; i++) t.v[4][i][i][0] = R3;         // p4 (1,1,0)
  // p5 (1,1,2): l1 basis (y,z,x); l2 basis (xy,yz,z2,xz,x2-y2)
  t.v[5][0][2][0] = A;  t.v[5][2][0][0] = A;
  t.v[5][0][1][1] = A;  t.v[5][1][0][1] = A;
  t.v[5][1][1][2] = B2; t.v[5][0][0][2] = -B1; t.v[5][2][2][2] = -B1;
  t.v[5][1][2][3] = A;  t.v[5][2][1][3] = A;
  t.v[5][2][2][4] = A;  t.v[5][0][0][4] = -A;
  // p6 (1,2,1), p8 (2,1,1): permutations of p5 (even l-sum => symmetric)
  for (int i = 0; i < 3; i++)
    for (int j = 0; j < 3; j++)
      for (int k = 0; k < 5; k++) {
        t.v[6][i][k][j] = t.v[5][i][j][k];
        t.v[8][k][i][j] = t.v[5][i][j][k];
      }
  for (int i = 0; i < 5; i++) t.v[7][i][0][i] = R5;         // p7 (2,0,2)
  for (int i = 0; i < 5; i++) t.v[9][i][i][0] = R5;         // p9 (2,2,0)
  // p10 (2,2,2): Tr(B_a B_b B_c) construction, normalized
  sym3(t, 10, 2, 2, 2,  C1);
  sym3(t, 10, 0, 0, 2, -C1);
  sym3(t, 10, 4, 4, 2, -C1);
  sym3(t, 10, 1, 1, 2,  C2);
  sym3(t, 10, 3, 3, 2,  C2);
  sym3(t, 10, 0, 1, 3,  C3);
  sym3(t, 10, 1, 1, 4, -C3);
  sym3(t, 10, 3, 3, 4,  C3);
  return t;
}

__constant__ CGTab cCG = buildCG();

__device__ constexpr int PI1[11]  = {0,0,0,1,1,1,1,2,2,2,2};
__device__ constexpr int PI2[11]  = {0,1,2,0,1,1,2,0,1,2,2};
__device__ constexpr int PI3[11]  = {0,1,2,1,0,2,1,2,1,0,2};
__device__ constexpr int IRD[3]   = {1,3,5};      // 2l+1
__device__ constexpr int IRO[3]   = {0,64,256};   // feature offsets
__device__ constexpr int ACC0[3]  = {0,1,4};      // accumulator base per i3

// ---- one-time ws f32[p][u][v][w] -> f16 wsH[p][u][w][v] (WMMA-A friendly) --
__global__ __launch_bounds__(256) void w_cvt_f16_T(const float* __restrict__ ws,
                                                   _Float16* __restrict__ wsH,
                                                   int total) {
  int e = blockIdx.x * 256 + threadIdx.x;
  if (e >= total) return;
  int pu = e >> 12;          // p*64+u
  int r  = e & 4095;         // v*64+w
  int v  = r >> 6;
  int w  = r & 63;
  wsH[(pu << 12) + (w << 6) + v] = (_Float16)ws[e];
}

// ---------------------------------------------------------------------------
template<bool PRE>
__global__ __launch_bounds__(128) void tp_wmma(const float* __restrict__ x1,
                                               const float* __restrict__ x2,
                                               const float* __restrict__ wsF,
                                               const _Float16* __restrict__ wsH,
                                               float* __restrict__ out, int B) {
  __shared__ float sX1[16 * 576];
  __shared__ float sX2[16 * 576];
  __shared__ __align__(16) _Float16 sW[2 * 64 * 64];  // ping-pong [w][v] (A)
  __shared__ __align__(16) _Float16 sZ[80 * 64];      // [c][v], c=b+16k (B)
  __shared__ float sT[16 * 32];                       // t[b][j*d3+k]

  const int tid  = threadIdx.x;
  const int lane = tid & 31;
  const int wv   = tid >> 5;     // wave id = w-tile (rows 16*wv .. 16*wv+15)
  const int col  = lane & 15;    // N index / A-row index
  const int hi   = lane >> 4;    // lane half (selects K sub-group per layout)
  const int bb   = blockIdx.x * 16;

  { // stage batch tile of x1/x2 (contiguous 9216 floats each)
    const float4* g1 = (const float4*)(x1 + (size_t)bb * 576);
    const float4* g2 = (const float4*)(x2 + (size_t)bb * 576);
    float4* s1 = (float4*)sX1;
    float4* s2 = (float4*)sX2;
    for (int i = tid; i < 2304; i += 128) { s1[i] = g1[i]; s2[i] = g2[i]; }
  }

  v8f acc[9] = {};   // 1 tile (l=0) + 3 tiles (l=1) + 5 tiles (l=2)

  #pragma unroll
  for (int p = 0; p < 11; p++) {
    const int i1 = PI1[p], i2 = PI2[p], i3 = PI3[p];
    const int d1 = IRD[i1], d2 = IRD[i2], d3 = IRD[i3];
    const int o1 = IRO[i1], o2 = IRO[i2];
    const int ab = ACC0[i3];

#if HAVE_TDM
    if (PRE && wv == 0)   // prime the pipeline: DMA slice u=0 into buffer 0
      tdm_load_w(wsH + (((size_t)p * 64) << 12),
                 (unsigned int)(uintptr_t)&sW[0]);
#endif

    for (int u = 0; u < 64; u++) {
      const int buf = (u & 1) << 12;
      __syncthreads();   // sW[buf]/sZ/sT free for reuse

      // ---- stage W slice (p,u): sW[buf][w][v] in f16 ---------------------
      if (PRE) {
#if HAVE_TDM
        if (wv == 0 && u < 63)   // DMA next slice into the other buffer
          tdm_load_w(wsH + (((size_t)p * 64 + u + 1) << 12),
                     (unsigned int)(uintptr_t)&sW[(~u & 1) << 12]);
#else
        const ulonglong2* g =
            (const ulonglong2*)(wsH + (((size_t)p * 64 + u) << 12));
        ulonglong2* s = (ulonglong2*)&sW[buf];
        #pragma unroll
        for (int i = 0; i < 4; i++) s[tid + i * 128] = g[tid + i * 128];
        if (u < 63)
          __builtin_prefetch(
              (const void*)(wsH + (((size_t)p * 64 + u + 1) << 12) + tid * 32),
              0, 1);
#endif
      } else {
        const float4* g = (const float4*)(wsF + (((size_t)p * 64 + u) << 12));
        #pragma unroll
        for (int i = 0; i < 8; i++) {
          float4 f = g[tid + i * 128];
          int e = (tid + i * 128) << 2;   // element = v*64 + w
          int v = e >> 6, w0 = e & 63;
          sW[buf + (w0 + 0) * 64 + v] = (_Float16)f.x;
          sW[buf + (w0 + 1) * 64 + v] = (_Float16)f.y;
          sW[buf + (w0 + 2) * 64 + v] = (_Float16)f.z;
          sW[buf + (w0 + 3) * 64 + v] = (_Float16)f.w;
        }
      }

      // ---- t[b][j,k] = sum_i C[i,j,k] * x1[b,u,i] ------------------------
      const int nT = 16 * d2 * d3;
      for (int idx = tid; idx < nT; idx += 128) {
        int b = idx / (d2 * d3), jk = idx - b * (d2 * d3);
        int j = jk / d3, k = jk - j * d3;
        float s = 0.f;
        #pragma unroll
        for (int i = 0; i < d1; i++)
          s += cCG.v[p][i][j][k] * sX1[b * 576 + o1 + u * d1 + i];
        sT[b * 32 + jk] = s;
      }
      __syncthreads();

      // ---- Z[(b,v),k] -> sZ[b+16k][v], packed 2xf16 stores ---------------
      #pragma unroll
      for (int r = 0; r < 4; r++) {
        int idx = tid + (r << 7);          // 16 b * 32 v-pairs
        int b = idx >> 5, vp = idx & 31;   // v = 2*vp, 2*vp+1
        const float* xb = &sX2[b * 576 + o2 + (vp << 1) * d2];
        const float* tb = &sT[b * 32];
        #pragma unroll
        for (int k = 0; k < d3; k++) {
          float z0 = 0.f, z1 = 0.f;
          #pragma unroll
          for (int j = 0; j < d2; j++) {
            float tj = tb[j * d3 + k];
            z0 += tj * xb[j];
            z1 += tj * xb[d2 + j];
          }
          h2v pk;
          pk[0] = (_Float16)z0;
          pk[1] = (_Float16)z1;
          *(h2v*)&sZ[((b + (k << 4)) << 6) + (vp << 1)] = pk;
        }
      }

#if HAVE_TDM
      if (PRE && wv == 0) {  // slice u must be resident before WMMA barrier
        if (u < 63) asm volatile("s_wait_tensorcnt 0x1" ::: "memory");
        else        asm volatile("s_wait_tensorcnt 0x0" ::: "memory");
      }
#endif
      __syncthreads();

      // ---- WMMA: acc(16w x 16cols) += A(16w x 64v) * B(64v x 16cols) -----
      // A fragment (16-bit A 16x32): lanes 0-15 hold K 0-7 & 16-23,
      // lanes 16-31 hold K 8-15 & 24-31.  B fragment (16-bit B 32x16):
      // same K layout as A (lanes 0-15 K 0-7 & 16-23, 16-31 K 8-15 & 24-31).
      const _Float16* wrow = &sW[buf + ((wv * 16 + col) << 6)];
      #pragma unroll
      for (int ks = 0; ks < 2; ks++) {     // K = 64 per u-step -> 2 wmma
        const int ka = ks * 32 + (hi ? 8 : 0);
        v8h a0 = *(const v8h*)(wrow + ka);
        v8h a1 = *(const v8h*)(wrow + ka + 16);
        v16h Af;
        #pragma unroll
        for (int i = 0; i < 8; i++) { Af[i] = a0[i]; Af[i + 8] = a1[i]; }
        const int kb = ks * 32 + (hi ? 8 : 0);
        #pragma unroll
        for (int k = 0; k < d3; k++) {
          const _Float16* zc = &sZ[((((k << 4) + col)) << 6) + kb];
          v8h b0 = *(const v8h*)(zc);
          v8h b1 = *(const v8h*)(zc + 16);
          v16h Bf;
          #pragma unroll
          for (int i = 0; i < 8; i++) { Bf[i] = b0[i]; Bf[i + 8] = b1[i]; }
          acc[ab + k] = __builtin_amdgcn_wmma_f32_16x16x32_f16(
              false, Af, false, Bf, (short)0, acc[ab + k], false, false);
        }
      }
    } // u
  } // p

  // ---- writeback: D layout VGPR g -> M = g (lanes 0-15) / 8+g (16-31) ----
  const int b = bb + col;
  if (b < B) {
    float* ob = out + (size_t)b * 576;
    #pragma unroll
    for (int i3 = 0; i3 < 3; i3++) {
      const int d3 = IRD[i3], o3 = IRO[i3], ab = ACC0[i3];
      #pragma unroll
      for (int k = 0; k < d3; k++) {
        v8f c = acc[ab + k];
        #pragma unroll
        for (int g = 0; g < 8; g++) {
          int w = wv * 16 + (hi ? 8 : 0) + g;
          ob[o3 + w * d3 + k] = c[g];
        }
      }
    }
  }
}

// ---------------------------------------------------------------------------
extern "C" void kernel_launch(void* const* d_in, const int* in_sizes, int n_in,
                              void* d_out, int out_size, void* d_ws,
                              size_t ws_size, hipStream_t stream) {
  const float* x1 = (const float*)d_in[0];
  const float* x2 = (const float*)d_in[1];
  const float* ws = (const float*)d_in[2];
  float* out = (float*)d_out;
  const int B = in_sizes[0] / 576;
  const int nblk = (B + 15) / 16;

  const int wtotal = 11 * 64 * 64 * 64;                  // 2,883,584
  const size_t need = (size_t)wtotal * sizeof(_Float16); // 5.75 MB
  if (ws_size >= need) {
    _Float16* wsH = (_Float16*)d_ws;
    w_cvt_f16_T<<<(wtotal + 255) / 256, 256, 0, stream>>>(ws, wsH, wtotal);
    tp_wmma<true><<<nblk, 128, 0, stream>>>(x1, x2, ws, wsH, out, B);
  } else {
    tp_wmma<false><<<nblk, 128, 0, stream>>>(x1, x2, ws, nullptr, out, B);
  }
}
